// XFADS_10350871184024
// MI455X (gfx1250) — compile-verified, matrix operand order checked by reading, not run on hardware
//
#include <hip/hip_runtime.h>
#include <hip/hip_bf16.h>

// ---- dims ----
#define B_  128
#define T_  1000
#define N_  200
#define D_  64
#define H_  256
#define U_  8

typedef __attribute__((ext_vector_type(16))) _Float16 v16h;
typedef __attribute__((ext_vector_type(8)))  float    v8f;

union frag_u { uint4 u[2]; v16h h; };
union h4_u   { _Float16 h[4]; unsigned long long u; };
union h8_u   { _Float16 h[8]; uint4 u; };

// ---------------- WMMA fragment helpers (16x16x32 f16 layouts) ----------------
// A (16xK tile): lane L<16 holds row L, K = k0+{0..7,16..23}; lane L>=16 holds
// row L-16, K = k0+{8..15,24..31}. Both 8-half runs are 16B-aligned -> 2x b128.
__device__ __forceinline__ v16h frag_A(const _Float16* p, int ld, int row0, int k0) {
    const int L = threadIdx.x & 31;
    const _Float16* r = p + (size_t)(row0 + (L & 15)) * ld + k0 + ((L >> 4) << 3);
    frag_u f;
    f.u[0] = *(const uint4*)(r);
    f.u[1] = *(const uint4*)(r + 16);
    return f.h;
}

// B fragments from pre-swizzled ("packed") weights: lane L's 16 halves for
// tile (kk,nt) are contiguous at ((kk*ntiles+nt)*32 + L) * 16 halves.
__device__ __forceinline__ v16h frag_Bp(const _Float16* packed, int ntiles,
                                        int kk, int nt) {
    const int L = threadIdx.x & 31;
    const uint4* q =
        (const uint4*)(packed + (((size_t)(kk * ntiles + nt) * 32 + L) << 4));
    frag_u f;
    f.u[0] = q[0];
    f.u[1] = q[1];
    return f.h;
}

__device__ __forceinline__ v8f wmma_f16(v16h a, v16h b, v8f c) {
    return __builtin_amdgcn_wmma_f32_16x16x32_f16(false, a, false, b, (short)0, c,
                                                  false, false);
}

// Branch-free tanh: 1 - 2*rcp(e^{2x}+1).  e->inf => 1, e->0 => -1.
__device__ __forceinline__ float fast_tanh(float x) {
    float e = __expf(2.f * x);
    return 1.f - 2.f * __builtin_amdgcn_rcpf(e + 1.f);
}

__device__ __forceinline__ float constrain_j(float x) {   // -softplus(x) - 1e-3
    float sp = (x > 20.f) ? x : __logf(1.f + __expf(x));
    return -sp - 1e-3f;
}

// ------------ kernel 0: f32 -> f16 pad + fragment-pack a weight matrix -------
// dst element idx: tile = idx/512 (tile = kk*ntiles+nt), lane = (idx/16)%32,
// i = idx%16. Source (k,n): k = kk*32 + 8*(lane>=16) + (i<8 ? i : i+8),
// n = nt*16 + (lane&15).
__global__ void __launch_bounds__(256) k_pack(const float* __restrict__ src,
                                              _Float16* __restrict__ dst,
                                              int K, int N, int Kp) {
    int idx = blockIdx.x * 256 + threadIdx.x;
    if (idx >= Kp * N) return;
    int ntiles = N >> 4;
    int tile = idx >> 9;
    int rem  = idx & 511;
    int L = rem >> 4, i = rem & 15;
    int kk = tile / ntiles, nt = tile - kk * ntiles;
    int k = kk * 32 + ((L >> 4) << 3) + ((i < 8) ? i : (i + 8));
    int n = nt * 16 + (L & 15);
    dst[idx] = (k < K) ? (_Float16)src[k * N + n] : (_Float16)0.f;
}

// ---------------- kernel 1: alpha encoder ----------------------------------
__global__ void __launch_bounds__(256) k_alpha(const float* __restrict__ y,
                                               const _Float16* __restrict__ W1p,
                                               const float* __restrict__ b1,
                                               const _Float16* __restrict__ W2p,
                                               const float* __restrict__ b2,
                                               float* __restrict__ aF32) {
    extern __shared__ __align__(16) char smem[];
    _Float16* yT  = (_Float16*)smem;           // 128 x 224
    _Float16* hT  = yT + 128 * 224;            // 128 x 256
    int*      msk = (int*)(hT + 128 * 256);    // 128
    const int tid = threadIdx.x;
    const size_t rbase = (size_t)blockIdx.x * 128;

    if (tid < 128) msk[tid] = 1;
    // zero-pad cols 200..223 (6 x u64 per row)
    for (int i = tid; i < 128 * 6; i += 256) {
        int row = i / 6, q = i - row * 6;
        *(unsigned long long*)(yT + row * 224 + 200 + q * 4) = 0ull;
    }
    __syncthreads();
    // vectorized y staging: 128 rows x 50 float4 -> 4x f16 u64 stores
#pragma unroll 5
    for (int i = 0; i < 25; ++i) {             // 6400 float4 / 256
        int idx = tid + i * 256;
        int row = idx / 50, c4 = idx - row * 50;
        float4 v = *(const float4*)(y + (rbase + row) * N_ + c4 * 4);
        unsigned bad = (__float_as_uint(v.x) & 0x7f800000u) == 0x7f800000u;
        bad |= (__float_as_uint(v.y) & 0x7f800000u) == 0x7f800000u;
        bad |= (__float_as_uint(v.z) & 0x7f800000u) == 0x7f800000u;
        bad |= (__float_as_uint(v.w) & 0x7f800000u) == 0x7f800000u;
        if (bad) { v = make_float4(0.f, 0.f, 0.f, 0.f); atomicAnd(&msk[row], 0); }
        h4_u w;
        w.h[0] = (_Float16)v.x; w.h[1] = (_Float16)v.y;
        w.h[2] = (_Float16)v.z; w.h[3] = (_Float16)v.w;
        *(unsigned long long*)(yT + row * 224 + c4 * 4) = w.u;
    }
    __syncthreads();

    const int wave = tid >> 5, lane = tid & 31;
    const int row0 = wave * 16;                // 8 waves x 16 rows

    // GEMM1: hidden = tanh(yT @ W1 + b1)  (K=224 -> 7 k-steps, 16 n-tiles)
    v16h af[7];
#pragma unroll
    for (int kk = 0; kk < 7; ++kk) af[kk] = frag_A(yT, 224, row0, kk * 32);
    for (int nt = 0; nt < 16; ++nt) {
        v8f acc = {};
#pragma unroll
        for (int kk = 0; kk < 7; ++kk)
            acc = wmma_f16(af[kk], frag_Bp(W1p, 16, kk, nt), acc);
        int col = nt * 16 + (lane & 15);
        float bias = b1[col];
#pragma unroll
        for (int v = 0; v < 8; ++v) {
            int r = row0 + ((lane >> 4) << 3) + v;
            hT[r * H_ + col] = (_Float16)fast_tanh(acc[v] + bias);
        }
    }
    __syncthreads();

    // GEMM2: a = constrain(hT @ W2 + b2) * mask   (K=256 -> 8 k-steps, 8 n-tiles)
    v16h hf[8];
#pragma unroll
    for (int kk = 0; kk < 8; ++kk) hf[kk] = frag_A(hT, H_, row0, kk * 32);
    for (int nt = 0; nt < 8; ++nt) {
        v8f acc = {};
#pragma unroll
        for (int kk = 0; kk < 8; ++kk)
            acc = wmma_f16(hf[kk], frag_Bp(W2p, 8, kk, nt), acc);
        int col = nt * 16 + (lane & 15);
        float bias = b2[col];
#pragma unroll
        for (int v = 0; v < 8; ++v) {
            int r = row0 + ((lane >> 4) << 3) + v;
            float z = acc[v] + bias;
            if (col >= D_) z = constrain_j(z);
            z *= (float)msk[r];
            aF32[(rbase + r) * (2 * D_) + col] = z;
        }
    }
}

// ---------------- kernel 2: beta backward RNN ------------------------------
__global__ void __launch_bounds__(256) k_beta(const float* __restrict__ aF32,
                                              const _Float16* __restrict__ Wcat,
                                              const float* __restrict__ bb,
                                              const float* __restrict__ bbo,
                                              float* __restrict__ bF32) {
    extern __shared__ __align__(16) char smem[];
    _Float16* Wx = (_Float16*)smem;        // packed 128x256
    _Float16* Wh = Wx + 128 * 256;         // packed 256x256
    _Float16* Wo = Wh + 256 * 256;         // packed 256x128
    _Float16* h0 = Wo + 256 * 128;         // 16 x 256
    _Float16* h1 = h0 + 16 * 256;          // 16 x 256
    _Float16* aT = h1 + 16 * 256;          // 16 x 128
    const int tid = threadIdx.x;
    const int b0 = blockIdx.x * 16;

    {   // stage Wx|Wh|Wo (contiguous packed in ws): 262144 B = 16384 uint4
        const uint4* s = (const uint4*)Wcat;
        uint4* d = (uint4*)Wx;
        for (int i = tid; i < 16384; i += 256) d[i] = s[i];
    }
    for (int i = tid; i < 16 * 256 / 8; i += 256)
        ((uint4*)h0)[i] = make_uint4(0, 0, 0, 0);
    __syncthreads();

    const int wave = tid >> 5, lane = tid & 31;
    const int lrow = tid >> 4, c8 = (tid & 15) * 8;  // 16 rows x 16 chunks of 8 f32
    // loop-invariant biases
    const float biasz0 = bb[wave * 32 + (lane & 15)];
    const float biasz1 = bb[wave * 32 + 16 + (lane & 15)];
    const float biaso  = bbo[wave * 16 + (lane & 15)];
    _Float16 *hcur = h0, *hnew = h1;

    for (int t = T_ - 1; t >= 0; --t) {
        {   // load a_t tile f32 -> f16 LDS (2x b128 load + 1x b128 store / thread)
            const float* src = aF32 + ((size_t)(b0 + lrow) * T_ + t) * (2 * D_) + c8;
            float4 x0 = *(const float4*)(src);
            float4 x1 = *(const float4*)(src + 4);
            h8_u w;
            w.h[0] = (_Float16)x0.x; w.h[1] = (_Float16)x0.y;
            w.h[2] = (_Float16)x0.z; w.h[3] = (_Float16)x0.w;
            w.h[4] = (_Float16)x1.x; w.h[5] = (_Float16)x1.y;
            w.h[6] = (_Float16)x1.z; w.h[7] = (_Float16)x1.w;
            *(uint4*)(aT + lrow * 128 + c8) = w.u;
            if (t > 0)
                __builtin_prefetch(aF32 + ((size_t)(b0 + lrow) * T_ + (t - 1)) * (2 * D_), 0, 1);
        }
        __syncthreads();

        v16h afr[4];
#pragma unroll
        for (int kk = 0; kk < 4; ++kk) afr[kk] = frag_A(aT, 128, 0, kk * 32);
        v16h hfr[8];
#pragma unroll
        for (int kk = 0; kk < 8; ++kk) hfr[kk] = frag_A(hcur, 256, 0, kk * 32);

#pragma unroll
        for (int j = 0; j < 2; ++j) {          // 16 n-tiles / 8 waves
            int nt = wave * 2 + j;
            v8f acc = {};
#pragma unroll
            for (int kk = 0; kk < 4; ++kk)
                acc = wmma_f16(afr[kk], frag_Bp(Wx, 16, kk, nt), acc);
#pragma unroll
            for (int kk = 0; kk < 8; ++kk)
                acc = wmma_f16(hfr[kk], frag_Bp(Wh, 16, kk, nt), acc);
            int col = nt * 16 + (lane & 15);
            float bias = j ? biasz1 : biasz0;
#pragma unroll
            for (int v = 0; v < 8; ++v) {
                int r = ((lane >> 4) << 3) + v;
                hnew[r * 256 + col] = (_Float16)fast_tanh(acc[v] + bias);
            }
        }
        __syncthreads();

        {   // out = constrain(hnew @ Wo + bbo)
            v16h nfr[8];
#pragma unroll
            for (int kk = 0; kk < 8; ++kk) nfr[kk] = frag_A(hnew, 256, 0, kk * 32);
            int nt = wave;                     // 8 n-tiles -> cols 0..127
            v8f acc = {};
#pragma unroll
            for (int kk = 0; kk < 8; ++kk)
                acc = wmma_f16(nfr[kk], frag_Bp(Wo, 8, kk, nt), acc);
            int col = nt * 16 + (lane & 15);
#pragma unroll
            for (int v = 0; v < 8; ++v) {
                int r = ((lane >> 4) << 3) + v;
                float z = acc[v] + biaso;
                if (col >= D_) z = constrain_j(z);
                bF32[((size_t)(b0 + r) * T_ + t) * (2 * D_) + col] = z;
            }
        }
        _Float16* tmp = hcur; hcur = hnew; hnew = tmp;
        __syncthreads();
    }
}

// ---------------- kernel 3: forward natural-parameter filter ---------------
__global__ void __launch_bounds__(256) k_filter(const float* __restrict__ aF32,
                                                const float* __restrict__ bF32,
                                                const float* __restrict__ u,
                                                const _Float16* __restrict__ Wcat,
                                                const float* __restrict__ fb1,
                                                const float* __restrict__ fb2,
                                                float* __restrict__ out) {
    extern __shared__ __align__(16) char smem[];
    _Float16* W1  = (_Float16*)smem;         // packed 96x256
    _Float16* W2  = W1 + 96 * 256;           // packed 256x64
    _Float16* xT  = W2 + 256 * 64;           // 16 x 96  ([m;u;pad])
    _Float16* hid = xT + 16 * 96;            // 16 x 256
    float* mN = (float*)(hid + 16 * 256);    // 16 x 64
    float* vN = mN + 1024;
    float* hp = vN + 1024;
    float* jp = hp + 1024;
    const int tid = threadIdx.x;
    const int b0 = blockIdx.x * 16;

    {   // stage fW1|fW2 (contiguous packed): 81920 B = 5120 uint4
        const uint4* s = (const uint4*)Wcat;
        uint4* d = (uint4*)W1;
        for (int i = tid; i < 5120; i += 256) d[i] = s[i];
    }
    for (int i = tid; i < 1024; i += 256) { hp[i] = 0.f; jp[i] = -0.5f; }
    for (int i = tid; i < 16 * 6; i += 256) {   // static zero pad cols 72..95
        int r = i / 6, q = i - r * 6;
        *(unsigned long long*)(xT + r * 96 + 72 + q * 4) = 0ull;
    }
    __syncthreads();

    const int wave = tid >> 5, lane = tid & 31;
    const float bias1a = fb1[wave * 32 + (lane & 15)];
    const float bias1b = fb1[wave * 32 + 16 + (lane & 15)];
    const float bias2  = (wave < 4) ? fb2[wave * 16 + (lane & 15)] : 0.f;

    for (int t = 0; t < T_; ++t) {
        // posterior update (elementwise over 16x64 states)
#pragma unroll
        for (int i = 0; i < 4; ++i) {
            int pid = tid + i * 256;
            int r = pid >> 6, d = pid & 63;
            size_t base = ((size_t)(b0 + r) * T_ + t) * (2 * D_);
            float abh = aF32[base + d] + bF32[base + d];
            float abj = aF32[base + D_ + d] + bF32[base + D_ + d];
            float hq = hp[pid] + abh;
            float jq = jp[pid] + abj;
            float vv = -0.5f * __builtin_amdgcn_rcpf(jq);
            float m = vv * hq;
            out[base + d]      = m;
            out[base + D_ + d] = vv;
            vN[pid] = vv + 0.1f;                 // predict variance
            xT[r * 96 + d] = (_Float16)m;
            if (t + 1 < T_) {
                __builtin_prefetch(&aF32[base + 2 * D_], 0, 1);
                __builtin_prefetch(&bF32[base + 2 * D_], 0, 1);
            }
        }
        if (tid < 128) {                          // control input slots 64..71
            int r = tid >> 3, j = tid & 7;
            xT[r * 96 + D_ + j] = (_Float16)u[((size_t)(b0 + r) * T_ + t) * U_ + j];
        }
        __syncthreads();

        // hidden = tanh(xT @ fW1 + fb1)  (K=96 -> 3 k-steps)
        v16h xfr[3];
#pragma unroll
        for (int kk = 0; kk < 3; ++kk) xfr[kk] = frag_A(xT, 96, 0, kk * 32);
#pragma unroll
        for (int j = 0; j < 2; ++j) {
            int nt = wave * 2 + j;
            v8f acc = {};
#pragma unroll
            for (int kk = 0; kk < 3; ++kk)
                acc = wmma_f16(xfr[kk], frag_Bp(W1, 16, kk, nt), acc);
            int col = nt * 16 + (lane & 15);
            float bias = j ? bias1b : bias1a;
#pragma unroll
            for (int v = 0; v < 8; ++v) {
                int r = ((lane >> 4) << 3) + v;
                hid[r * 256 + col] = (_Float16)fast_tanh(acc[v] + bias);
            }
        }
        __syncthreads();

        // m_next = hid @ fW2 + fb2  (4 n-tiles on waves 0..3)
        if (wave < 4) {
            v16h hfr[8];
#pragma unroll
            for (int kk = 0; kk < 8; ++kk) hfr[kk] = frag_A(hid, 256, 0, kk * 32);
            int nt = wave;
            v8f acc = {};
#pragma unroll
            for (int kk = 0; kk < 8; ++kk)
                acc = wmma_f16(hfr[kk], frag_Bp(W2, 4, kk, nt), acc);
            int col = nt * 16 + (lane & 15);
#pragma unroll
            for (int v = 0; v < 8; ++v) {
                int r = ((lane >> 4) << 3) + v;
                mN[r * 64 + col] = acc[v] + bias2;
            }
        }
        __syncthreads();

        // carry = (m_next / v_next, -0.5 / v_next)  via one rcp
#pragma unroll
        for (int i = 0; i < 4; ++i) {
            int pid = tid + i * 256;
            float rv = __builtin_amdgcn_rcpf(vN[pid]);
            hp[pid] = mN[pid] * rv;
            jp[pid] = -0.5f * rv;
        }
        __syncthreads();
    }
}

// ---------------- host: workspace layout + launches ------------------------
extern "C" void kernel_launch(void* const* d_in, const int* in_sizes, int n_in,
                              void* d_out, int out_size, void* d_ws, size_t ws_size,
                              hipStream_t stream) {
    const float* y   = (const float*)d_in[1];
    const float* u   = (const float*)d_in[2];
    const float* aW1 = (const float*)d_in[3];
    const float* ab1 = (const float*)d_in[4];
    const float* aW2 = (const float*)d_in[5];
    const float* ab2 = (const float*)d_in[6];
    const float* bWx = (const float*)d_in[7];
    const float* bWh = (const float*)d_in[8];
    const float* bb  = (const float*)d_in[9];
    const float* bWo = (const float*)d_in[10];
    const float* bbo = (const float*)d_in[11];
    const float* fW1 = (const float*)d_in[12];
    const float* fb1 = (const float*)d_in[13];
    const float* fW2 = (const float*)d_in[14];
    const float* fb2 = (const float*)d_in[15];
    float* out = (float*)d_out;

    char* ws = (char*)d_ws;
    // packed f16 weights (contiguous groups so scans can bulk-copy to LDS)
    _Float16* wA1h = (_Float16*)(ws + 0);        // 224x256 packed
    _Float16* wA2h = (_Float16*)(ws + 114688);   // 256x128 packed
    _Float16* wBxh = (_Float16*)(ws + 180224);   // 128x256 | 256x256 | 256x128
    _Float16* wBhh = (_Float16*)(ws + 245760);
    _Float16* wBoh = (_Float16*)(ws + 376832);
    _Float16* wF1h = (_Float16*)(ws + 442368);   // 96x256 | 256x64
    _Float16* wF2h = (_Float16*)(ws + 491520);
    float*    aF32 = (float*)(ws + 524288);                       // [B,T,128]
    float*    bF32 = (float*)(ws + 524288 + 65536000ull);         // [B,T,128]

    auto pack = [&](const float* s, _Float16* d, int K, int N, int Kp) {
        int tot = Kp * N;
        k_pack<<<(tot + 255) / 256, 256, 0, stream>>>(s, d, K, N, Kp);
    };
    pack(aW1, wA1h, 200, 256, 224);
    pack(aW2, wA2h, 256, 128, 256);
    pack(bWx, wBxh, 128, 256, 128);
    pack(bWh, wBhh, 256, 256, 256);
    pack(bWo, wBoh, 256, 128, 256);
    pack(fW1, wF1h,  72, 256,  96);
    pack(fW2, wF2h, 256,  64, 256);

    size_t lds1 = (size_t)(128 * 224 + 128 * 256) * 2 + 128 * 4;                 // 123392
    k_alpha<<<(B_ * T_) / 128, 256, lds1, stream>>>(y, wA1h, ab1, wA2h, ab2, aF32);

    size_t lds2 = (size_t)(128 * 256 + 256 * 256 + 256 * 128 + 2 * 16 * 256 + 16 * 128) * 2; // 282624
    k_beta<<<B_ / 16, 256, lds2, stream>>>(aF32, wBxh, bb, bbo, bF32);

    size_t lds3 = (size_t)(96 * 256 + 256 * 64 + 16 * 96 + 16 * 256) * 2 + 4 * 1024 * 4;     // 109568
    k_filter<<<B_ / 16, 256, lds3, stream>>>(aF32, bF32, u, wF1h, fb1, fb2, out);
}
